// Yolov8_82557861363909
// MI455X (gfx1250) — matrix operand order repaired and checked by reading, not compiled, not run on hardware
//
#include <hip/hip_runtime.h>
#include <hip/hip_bf16.h>

typedef __attribute__((ext_vector_type(16))) _Float16 v16h;
typedef __attribute__((ext_vector_type(8)))  _Float16 v8h;
typedef __attribute__((ext_vector_type(8)))  float    v8f;

__device__ __forceinline__ float silu_f(float v) {
    return v / (1.0f + __expf(-v));
}

// ---------------------------------------------------------------------------
// Weight repack: f32 OIHW [Co,Ci,KH,KW] -> f16 [Co][K], K = (ky*KW+kx)*Ci + ci
// ---------------------------------------------------------------------------
__global__ void repack_weights(const float* __restrict__ w, _Float16* __restrict__ wf,
                               int Co, int Ci, int KH, int KW) {
    int Ktot = Ci * KH * KW;
    int idx = blockIdx.x * blockDim.x + threadIdx.x;
    if (idx >= Co * Ktot) return;
    int co  = idx / Ktot;
    int k   = idx % Ktot;
    int ci  = k % Ci;
    int tap = k / Ci;
    int ky = tap / KW, kx = tap % KW;
    wf[idx] = (_Float16)w[(((size_t)co * Ci + ci) * KH + ky) * KW + kx];
}

// ---------------------------------------------------------------------------
// conv0: 3x3 s2 p1, 3->32, f32 NCHW in -> SiLU -> f16 NHWC out [N,320,320,32]
// K=27 too skinny for 16x16x32 WMMA; VALU FMA (1.4 GF of 28 total).
// ---------------------------------------------------------------------------
__global__ void conv0_silu(const float* __restrict__ x, const float* __restrict__ w,
                           const float* __restrict__ b, _Float16* __restrict__ out, int N) {
    const int OH = 320, OW = 320, IH = 640, IW = 640;
    int idx = blockIdx.x * blockDim.x + threadIdx.x;
    int total = N * OH * OW * 32;
    if (idx >= total) return;
    int c = idx & 31;
    int t = idx >> 5;
    int ox = t % OW; t /= OW;
    int oy = t % OH; t /= OH;
    int n = t;
    float acc = b[c];
    #pragma unroll
    for (int ci = 0; ci < 3; ++ci)
        #pragma unroll
        for (int ky = 0; ky < 3; ++ky) {
            int iy = oy * 2 - 1 + ky;
            if (iy < 0 || iy >= IH) continue;
            #pragma unroll
            for (int kx = 0; kx < 3; ++kx) {
                int ix = ox * 2 - 1 + kx;
                if (ix < 0 || ix >= IW) continue;
                acc += x[(((size_t)n * 3 + ci) * IH + iy) * IW + ix] *
                       w[((c * 3 + ci) * 3 + ky) * 3 + kx];
            }
        }
    out[idx] = (_Float16)silu_f(acc);
}

// ---------------------------------------------------------------------------
// Implicit-GEMM conv via v_wmma_f32_16x16x32_f16 (wave32), N-blocked x4.
// Block = 4 waves; weight tile [NTILE x KTOT] staged in LDS once per block;
// each wave: 16 pixels x NTILE out-channels -> NB WMMAs per A fetch.
// A (16xK):  lane<16: K {0..15}; lane>=16: K {16..31}
// B (Kx16):  lane&15 = column, lane>>4 = K-half; [co][K] rows in LDS give a
//            contiguous 32B ds read per lane.
// C/D:       column = lane&15 (constant per lane), row = vgpr + 8*(lane>>4)
// ---------------------------------------------------------------------------
template<int CIN, int COUT, int KH, int KW, int STRIDE, int PAD,
         int IPITCH, int IOFF, int RPITCH, int ROFF, bool RESIDUAL, bool OUT_F32_NCHW>
__global__ __launch_bounds__(128)
void conv_wmma(const _Float16* __restrict__ in, const _Float16* __restrict__ wf,
               const float* __restrict__ bias, const _Float16* __restrict__ res,
               _Float16* __restrict__ outh, float* __restrict__ outf,
               int N, int IH, int IW, int OH, int OW) {
    constexpr int KTOT   = KH * KW * CIN;
    constexpr int KSTEPS = KTOT / 32;               // CIN is a multiple of 32
    constexpr int NTILE  = (COUT < 64) ? COUT : 64; // out-channels per block
    constexpr int NB     = NTILE / 16;              // WMMA N-tiles per wave

    __shared__ __align__(32) _Float16 wlds[NTILE * KTOT];

    // Cooperative stage of the weight tile into LDS (16B chunks).
    {
        const uint4* src = (const uint4*)(wf + (size_t)blockIdx.y * NTILE * KTOT);
        uint4* dst = (uint4*)wlds;
        const int nchunk = NTILE * KTOT / 8;
        for (int i = threadIdx.x; i < nchunk; i += blockDim.x)
            dst[i] = src[i];
    }
    __syncthreads();

    const int lane = threadIdx.x & 31;
    const int wv   = threadIdx.x >> 5;
    const int tilesPerRow = OW >> 4;
    const int totalM = N * OH * tilesPerRow;
    const int mtile = blockIdx.x * (blockDim.x >> 5) + wv;   // wave-uniform
    if (mtile >= totalM) return;

    int txi  = mtile % tilesPerRow;
    int rest = mtile / tilesPerRow;
    int oy   = rest % OH;
    int nimg = rest / OH;
    int oxbase = txi << 4;

    const int col = lane & 15;     // B/C/D column
    const int hi  = lane >> 4;     // K-half select
    const int m   = lane & 15;     // A row (pixel within tile)
    const int ox  = oxbase + m;

    v8f acc[NB] = {};

    for (int ks = 0; ks < KSTEPS; ++ks) {
        const int k0  = ks * 32;
        const int tap = k0 / CIN;
        const int cw  = k0 % CIN;
        const int ky  = tap / KW, kx = tap % KW;
        const int iy  = oy * STRIDE - PAD + ky;
        const int ix  = ox * STRIDE - PAD + kx;
        bool inb = (PAD == 0) || ((iy >= 0) & (iy < IH) & (ix >= 0) & (ix < IW));

        v8h c0 = {}; v8h c1 = {};
        if (inb) {
            const _Float16* p = in + ((size_t)(nimg * IH + iy) * IW + ix) * IPITCH
                                   + IOFF + cw + hi * 16;
            c0 = *(const v8h*)p;          // K {0..7} or {16..23}
            c1 = *(const v8h*)(p + 8);    // K {8..15} or {24..31}
        }
        v16h a = __builtin_shufflevector(c0, c1, 0, 1, 2, 3, 4, 5, 6, 7,
                                                 8, 9, 10, 11, 12, 13, 14, 15);
        #pragma unroll
        for (int j = 0; j < NB; ++j) {
            v16h bm = *(const v16h*)(&wlds[(j * 16 + col) * KTOT + k0 + hi * 16]);
            acc[j] = __builtin_amdgcn_wmma_f32_16x16x32_f16(
                         false, a, false, bm, (short)0, acc[j], false, false);
        }
    }

    #pragma unroll
    for (int j = 0; j < NB; ++j) {
        const int co = blockIdx.y * NTILE + j * 16 + col;
        const float bv = bias[co];
        #pragma unroll
        for (int i = 0; i < 8; ++i) {
            const int mm  = i + hi * 8;            // C/D row
            const int oxm = oxbase + mm;
            float v = silu_f(acc[j][i] + bv);
            const size_t pix = (size_t)(nimg * OH + oy) * OW + oxm;
            if (RESIDUAL)
                v += (float)res[pix * RPITCH + ROFF + j * 16 + col];
            if (OUT_F32_NCHW)
                outf[(((size_t)nimg * COUT + co) * OH + oy) * OW + oxm] = v;
            else
                outh[pix * COUT + co] = (_Float16)v;
        }
    }
}

// cat = [y(ch0..63) | b2(ch0..31)] -> 96-pitch NHWC f16
__global__ void concat96(const _Float16* __restrict__ y, const _Float16* __restrict__ b2,
                         _Float16* __restrict__ cat, int total) {
    int idx = blockIdx.x * blockDim.x + threadIdx.x;
    if (idx >= total) return;
    int c = idx % 96;
    size_t pix = (size_t)(idx / 96);
    cat[idx] = (c < 64) ? y[pix * 64 + c] : b2[pix * 32 + (c - 64)];
}

// ---------------------------------------------------------------------------
extern "C" void kernel_launch(void* const* d_in, const int* in_sizes, int n_in,
                              void* d_out, int out_size, void* d_ws, size_t ws_size,
                              hipStream_t stream) {
    (void)in_sizes; (void)n_in; (void)out_size; (void)ws_size;
    const float* x   = (const float*)d_in[0];
    const float* w0  = (const float*)d_in[1];  const float* b0  = (const float*)d_in[2];
    const float* w1  = (const float*)d_in[3];  const float* b1  = (const float*)d_in[4];
    const float* wc0 = (const float*)d_in[5];  const float* bc0 = (const float*)d_in[6];
    const float* wb1 = (const float*)d_in[7];  const float* bb1 = (const float*)d_in[8];
    const float* wb2 = (const float*)d_in[9];  const float* bb2 = (const float*)d_in[10];
    const float* wc1 = (const float*)d_in[11]; const float* bc1 = (const float*)d_in[12];
    const float* w3  = (const float*)d_in[13]; const float* b3  = (const float*)d_in[14];
    float* out = (float*)d_out;

    const int N = 8;
    const size_t P320 = (size_t)N * 320 * 320;
    const size_t P160 = (size_t)N * 160 * 160;

    // -------- workspace carve-up (f16 buffers; ~131 MB peak, reuse regions) ----
    char* ws = (char*)d_ws;  size_t off = 0;
    auto take = [&](size_t bytes) -> char* {
        off = (off + 255) & ~(size_t)255;
        char* p = ws + off; off += bytes; return p;
    };
    _Float16* wf1  = (_Float16*)take((size_t)64  * 288 * 2);
    _Float16* wfc0 = (_Float16*)take((size_t)64  * 64  * 2);
    _Float16* wfb1 = (_Float16*)take((size_t)32  * 288 * 2);
    _Float16* wfb2 = (_Float16*)take((size_t)32  * 288 * 2);
    _Float16* wfc1 = (_Float16*)take((size_t)64  * 96  * 2);
    _Float16* wf3  = (_Float16*)take((size_t)128 * 576 * 2);
    _Float16* act0 = (_Float16*)take(P320 * 32 * 2);   // conv0 out; reused for cat(96ch)
    _Float16* act1 = (_Float16*)take(P160 * 64 * 2);   // conv1 out; reused for h2
    _Float16* ybuf = (_Float16*)take(P160 * 64 * 2);   // wc0 out (x1|x2)
    _Float16* b1b  = (_Float16*)take(P160 * 32 * 2);
    _Float16* b2b  = (_Float16*)take(P160 * 32 * 2);
    _Float16* cat  = act0;
    _Float16* h2   = act1;

    // -------- weight repack (f32 OIHW -> f16 [Co][K]) -------------------------
    auto rp = [&](const float* w, _Float16* wf, int Co, int Ci, int KH, int KW) {
        int tot = Co * Ci * KH * KW;
        repack_weights<<<(tot + 255) / 256, 256, 0, stream>>>(w, wf, Co, Ci, KH, KW);
    };
    rp(w1,  wf1,  64, 32, 3, 3);
    rp(wc0, wfc0, 64, 64, 1, 1);
    rp(wb1, wfb1, 32, 32, 3, 3);
    rp(wb2, wfb2, 32, 32, 3, 3);
    rp(wc1, wfc1, 64, 96, 1, 1);
    rp(w3,  wf3, 128, 64, 3, 3);

    // -------- stem -------------------------------------------------------------
    {   // conv0: 3->32 s2, VALU path, f16 NHWC out
        int tot = (int)(P320 * 32);
        conv0_silu<<<(tot + 255) / 256, 256, 0, stream>>>(x, w0, b0, act0, N);
    }
    // conv1: 32->64 s2  (M-tiles = 8*160*10 = 12800; 4 waves/block; y = 64/64)
    conv_wmma<32, 64, 3, 3, 2, 1, 32, 0, 0, 0, false, false>
        <<<dim3(3200, 1), 128, 0, stream>>>(act0, wf1, b1, nullptr, act1, nullptr,
                                            N, 320, 320, 160, 160);
    // -------- C2f --------------------------------------------------------------
    // wc0: 1x1 64->64
    conv_wmma<64, 64, 1, 1, 1, 0, 64, 0, 0, 0, false, false>
        <<<dim3(3200, 1), 128, 0, stream>>>(act1, wfc0, bc0, nullptr, ybuf, nullptr,
                                            N, 160, 160, 160, 160);
    // wb1: 3x3 32->32 on x2 = ybuf(pitch 64, offset 32)
    conv_wmma<32, 32, 3, 3, 1, 1, 64, 32, 0, 0, false, false>
        <<<dim3(3200, 1), 128, 0, stream>>>(ybuf, wfb1, bb1, nullptr, b1b, nullptr,
                                            N, 160, 160, 160, 160);
    // wb2: 3x3 32->32, fused residual add of x2 after SiLU
    conv_wmma<32, 32, 3, 3, 1, 1, 32, 0, 64, 32, true, false>
        <<<dim3(3200, 1), 128, 0, stream>>>(b1b, wfb2, bb2, ybuf, b2b, nullptr,
                                            N, 160, 160, 160, 160);
    // concat [x1|x2|b] -> 96ch
    {
        int tot = (int)(P160 * 96);
        concat96<<<(tot + 255) / 256, 256, 0, stream>>>(ybuf, b2b, cat, tot);
    }
    // wc1: 1x1 96->64
    conv_wmma<96, 64, 1, 1, 1, 0, 96, 0, 0, 0, false, false>
        <<<dim3(3200, 1), 128, 0, stream>>>(cat, wfc1, bc1, nullptr, h2, nullptr,
                                            N, 160, 160, 160, 160);
    // -------- downsample: w3 64->128 s2, f32 NCHW straight to d_out ------------
    // M-tiles = 8*80*5 = 3200; y = 128/64 = 2
    conv_wmma<64, 128, 3, 3, 2, 1, 64, 0, 0, 0, false, true>
        <<<dim3(800, 2), 128, 0, stream>>>(h2, wf3, b3, nullptr, nullptr, out,
                                           N, 160, 160, 80, 80);
}
